// VectorQuantizerEMA_25555055411348
// MI455X (gfx1250) — matrix operand order repaired and checked by reading, not compiled, not run on hardware
//
#include <hip/hip_runtime.h>
#include <hip/hip_bf16.h>

// VQ-VAE quantization for MI455X (gfx1250, wave32).
// inputs: [B=32, D=256, T=2048] fp32 ; embedding: [K=1024, D=256] fp32
// outputs (flat, in order): quantized [B,D,T] fp32, loss scalar, indices [N] (as float)

#define NUM_EMB   1024
#define EMB_DIM   256
#define BATCH     32
#define T_SZ      2048
#define N_ROWS    (BATCH * T_SZ)          // 65536 tokens
#define COMMIT_SCALE (0.5f / (float)((size_t)N_ROWS * EMB_DIM))

typedef __attribute__((ext_vector_type(16))) _Float16 v16h;
typedef __attribute__((ext_vector_type(8)))  float    v8f;
typedef __attribute__((ext_vector_type(4)))  unsigned v4u;
typedef __attribute__((ext_vector_type(8)))  int      v8i_t;
typedef __attribute__((ext_vector_type(4)))  int      v4i_t;

// ---------------------------------------------------------------------------
// Tensor Data Mover issue: one 3D descriptor moves a 16-code tile of BOTH the
// hi and lo f16 planes (planes adjacent in workspace, 512KB apart) into a
// contiguous 16KB LDS buffer.  D# layout per CDNA5 ISA.
//   data_size = 8B units; row = 512B = 64 units
//   tile: dim0=64, dim1=16 rows, dim2=2 planes
//   tensor_dim0_stride = 64 units; tensor_dim1_stride (plane) = 65536 units
// ---------------------------------------------------------------------------
#if __has_include(<hip/amd_detail/amd_gfx1250_TDM.h>)
  // amdgpu-toolchain (clang-23 / therock): 6-arg builtin
  #define TDM_CALL(g0, g1, g2, g3) \
      __builtin_amdgcn_tensor_load_to_lds((g0), (g1), (g2), (g3), (v8i_t)0, 0)
#elif __has_builtin(__builtin_amdgcn_tensor_load_to_lds)
  // ROCm 7.2 (clang-22): 5-arg builtin
  #define TDM_CALL(g0, g1, g2, g3) \
      __builtin_amdgcn_tensor_load_to_lds((g0), (g1), (g2), (g3), 0)
#else
  // portable fallback: raw TDM instruction, groups in SGPR tuples
  #define TDM_CALL(g0, g1, g2, g3) \
      asm volatile("tensor_load_to_lds %0, %1, %2, %3" \
                   :: "s"(g0), "s"(g1), "s"(g2), "s"(g3) : "memory")
#endif

__device__ __forceinline__ void tdm_issue_tile(unsigned lds_byte_addr,
                                               unsigned long long gaddr) {
    v4u g0;
    g0[0] = 1u;                                            // count=1 valid user D#
    g0[1] = lds_byte_addr;                                 // LDS dest (bytes)
    g0[2] = (unsigned)(gaddr & 0xFFFFFFFFull);             // global_addr[31:0]
    g0[3] = (unsigned)((gaddr >> 32) & 0x1FFFFFFull)       // global_addr[56:32]
            | (2u << 30);                                  // type=2 ("image")
    v8i_t g1;
    g1[0] = (int)(3u << 16);                               // data_size=3 (8B)
    g1[1] = (int)(64u << 16);                              // tensor_dim0 = 64 units
    g1[2] = (int)((unsigned)NUM_EMB << 16);                // tensor_dim1 = 1024 rows
    g1[3] = (int)(64u << 16);                              // tile_dim0 = 64 units
    g1[4] = (int)(16u | (2u << 16));                       // tile_dim1=16, tile_dim2=2
    g1[5] = 64;                                            // tensor_dim0_stride = 64
    g1[6] = 0;                                             // stride hi bits
    g1[7] = 1;                                             // tensor_dim1_stride=65536 (hi16)
    v4i_t g2;
    g2[0] = 2; g2[1] = 0; g2[2] = 0; g2[3] = 0;            // tensor_dim2 = 2 planes
    v4i_t g3;
    g3[0] = 0; g3[1] = 0; g3[2] = 0; g3[3] = 0;
    TDM_CALL(g0, g1, g2, g3);
}

// ---------------------------------------------------------------------------
// Prep: split codebook into f16 hi/lo planes and precompute 0.5*||e_k||^2.
// One block (256 threads) per code.
// ---------------------------------------------------------------------------
__global__ void vq_prep_kernel(const float* __restrict__ emb,
                               _Float16* __restrict__ e_hi,
                               _Float16* __restrict__ e_lo,
                               float* __restrict__ half_norm) {
    const int k = blockIdx.x;
    const int d = threadIdx.x;                       // 256 threads == EMB_DIM
    const float v = emb[k * EMB_DIM + d];
    const _Float16 hi = (_Float16)v;
    const _Float16 lo = (_Float16)(v - (float)hi);
    e_hi[k * EMB_DIM + d] = hi;
    e_lo[k * EMB_DIM + d] = lo;

    float s = v * v;
    #pragma unroll
    for (int off = 16; off > 0; off >>= 1) s += __shfl_xor(s, off, 32);

    __shared__ float red[8];
    const int wave = threadIdx.x >> 5;
    const int lane = threadIdx.x & 31;
    if (lane == 0) red[wave] = s;
    __syncthreads();
    if (threadIdx.x == 0) {
        float t = 0.f;
        #pragma unroll
        for (int i = 0; i < 8; ++i) t += red[i];
        half_norm[k] = 0.5f * t;
    }
}

// ---------------------------------------------------------------------------
// Main kernel. 256 threads = 8 wave32; each wave owns 16 tokens (A resident in
// VGPRs as f16 hi/lo fragments). Wave 0 double-buffers 16-code B tiles into
// LDS via the Tensor Data Mover (TENSORcnt). Score s = x.e - 0.5||e||^2,
// argmax == argmin distance. Grid: N_ROWS / 128 = 512 blocks.
// ---------------------------------------------------------------------------
__global__ __launch_bounds__(256, 1)
void vq_main_kernel(const float* __restrict__ x,        // [B, D, T]
                    const float* __restrict__ emb,      // [K, D] fp32 (exact gather)
                    const _Float16* __restrict__ e_hi,  // [K, D] (e_lo plane follows)
                    const float* __restrict__ half_norm,// [K]
                    float* __restrict__ out_q,          // [B, D, T]
                    float* __restrict__ out_loss,       // scalar (pre-zeroed)
                    float* __restrict__ out_idx) {      // [N] as float
    // [buf][hi|lo][code-in-tile][d] : 2*2*16*256 f16 = 32 KB.
    // TDM writes one buf (16KB: hi tile then lo tile, contiguous) per issue.
    __shared__ __attribute__((aligned(32))) _Float16 ldsB[2][2][16][EMB_DIM];
    __shared__ int   idx_lds[128];
    __shared__ float loss_red[8];

    const int tid  = threadIdx.x;
    const int wave = tid >> 5;
    const int lane = tid & 31;
    const int m    = lane & 15;          // A row (token) / B column (code) index
    const int hsel = lane >> 4;          // 0: lanes 0-15, 1: lanes 16-31

    const int blk      = blockIdx.x;         // 0..511
    const int b        = blk >> 4;           // batch index (16 blocks per b)
    const int t0_block = (blk & 15) << 7;    // 128 tokens per block
    const int t0       = t0_block + (wave << 4);
    const size_t bofs  = (size_t)b * EMB_DIM * T_SZ;

    // flat->LDS aperture: low 32 bits of a flat LDS pointer == LDS byte offset
    const unsigned ldsB_base = (unsigned)(unsigned long long)&ldsB[0][0][0][0];
    const unsigned long long ghi = (unsigned long long)e_hi;

    // Kick off tile 0 DMA before the (long) A-block load, for overlap.
    if (wave == 0) tdm_issue_tile(ldsB_base, ghi);

    // ---- Load this wave's 16x256 A block into registers as f16 hi/lo ------
    // f16 A-fragment k order this code uses (16x32 tile, same as B): lanes 0-15 hold M=lane,
    // element i -> K = i + 16*hsel.
    v16h a_hi[8], a_lo[8];
    {
        const int t = t0 + m;
        #pragma unroll
        for (int dc = 0; dc < 8; ++dc) {
            #pragma unroll
            for (int j = 0; j < 16; ++j) {
                const int kk = j + (hsel << 4);
                const int d  = dc * 32 + kk;
                const float v = x[bofs + (size_t)d * T_SZ + t];
                const _Float16 hi = (_Float16)v;
                a_hi[dc][j] = hi;
                a_lo[dc][j] = (_Float16)(v - (float)hi);
            }
        }
    }

    float best_s[8];
    int   best_i[8];
    #pragma unroll
    for (int r = 0; r < 8; ++r) { best_s[r] = -3.4e38f; best_i[r] = 0; }

    if (wave == 0) __builtin_amdgcn_s_wait_tensorcnt(0);  // tile 0 landed
    __syncthreads();

    // B-fragment k order this code uses (32x16 tile, same as A): lane holds column N=lane&15,
    // element i -> K = i + 16*hsel (16 contiguous halves -> 2x ds_load_b128).
    const int kofs = hsel << 4;

    for (int kt = 0; kt < 64; ++kt) {
        const int buf = kt & 1;
        // Issue DMA for next tile into the other buffer. Safe: the barrier that
        // ended the previous iteration guarantees no wave still reads buf^1.
        if (wave == 0 && kt + 1 < 64)
            tdm_issue_tile(ldsB_base + (unsigned)(buf ^ 1) * 16384u,
                           ghi + (unsigned long long)(kt + 1) * (16ull * 512ull));

        v8f acc = {};
        #pragma unroll
        for (int dc = 0; dc < 8; ++dc) {
            const v16h bh = *(const v16h*)&ldsB[buf][0][m][dc * 32 + kofs];
            const v16h bl = *(const v16h*)&ldsB[buf][1][m][dc * 32 + kofs];
            // (ah+al)*(bh+bl) ~= ah*bh + ah*bl + al*bh  (fp32 accumulate)
            acc = __builtin_amdgcn_wmma_f32_16x16x32_f16(false, a_hi[dc], false, bh,
                                                         (short)0, acc, false, false);
            acc = __builtin_amdgcn_wmma_f32_16x16x32_f16(false, a_hi[dc], false, bl,
                                                         (short)0, acc, false, false);
            acc = __builtin_amdgcn_wmma_f32_16x16x32_f16(false, a_lo[dc], false, bh,
                                                         (short)0, acc, false, false);
        }

        const int   code = kt * 16 + m;
        const float hn   = half_norm[code];
        // C layout: lanes 0-15 hold rows M=r (VGPR r), lanes 16-31 rows M=8+r.
        #pragma unroll
        for (int r = 0; r < 8; ++r) {
            const float s = acc[r] - hn;
            if (s > best_s[r]) { best_s[r] = s; best_i[r] = code; }
        }

        // Next tile's DMA complete (wave 0), then workgroup-wide handoff:
        // establishes tile kt+1 readiness AND read-protection for buf.
        if (wave == 0 && kt + 1 < 64) __builtin_amdgcn_s_wait_tensorcnt(0);
        __syncthreads();
    }

    // ---- Cross-lane argmax within each 16-lane half (tie -> lowest index) -
    #pragma unroll
    for (int r = 0; r < 8; ++r) {
        float s = best_s[r];
        int   i = best_i[r];
        #pragma unroll
        for (int off = 1; off < 16; off <<= 1) {
            const float os = __shfl_xor(s, off, 32);
            const int   oi = __shfl_xor(i, off, 32);
            if (os > s || (os == s && oi < i)) { s = os; i = oi; }
        }
        best_i[r] = i;
    }

    const int rowbase = blk * 128 + wave * 16;     // = b*T + t0
    if (lane == 0) {
        #pragma unroll
        for (int r = 0; r < 8; ++r) {
            idx_lds[wave * 16 + r] = best_i[r];
            out_idx[rowbase + r]   = (float)best_i[r];
        }
    } else if (lane == 16) {
        #pragma unroll
        for (int r = 0; r < 8; ++r) {
            idx_lds[wave * 16 + 8 + r] = best_i[r];
            out_idx[rowbase + 8 + r]   = (float)best_i[r];
        }
    }
    __syncthreads();

    // ---- Output + commitment loss: block covers [b, 0..255, t0_block+0..127]
    // Straight-through output is numerically exactly embedding[idx].
    const int tt  = tid & 127;
    const int idx = idx_lds[tt];
    float lsum = 0.f;
    for (int it = 0; it < 128; ++it) {
        const int d = (it << 1) + (tid >> 7);
        const size_t o = bofs + (size_t)d * T_SZ + t0_block + tt;
        const float e  = emb[idx * EMB_DIM + d];   // L2-resident gather
        const float xv = x[o];
        out_q[o] = e;
        const float diff = e - xv;
        lsum += diff * diff;
    }
    #pragma unroll
    for (int off = 16; off > 0; off >>= 1) lsum += __shfl_xor(lsum, off, 32);
    if (lane == 0) loss_red[wave] = lsum;
    __syncthreads();
    if (tid == 0) {
        float t = 0.f;
        #pragma unroll
        for (int i = 0; i < 8; ++i) t += loss_red[i];
        atomicAdd(out_loss, t * COMMIT_SCALE);
    }
}

// ---------------------------------------------------------------------------
extern "C" void kernel_launch(void* const* d_in, const int* in_sizes, int n_in,
                              void* d_out, int out_size, void* d_ws, size_t ws_size,
                              hipStream_t stream) {
    const float* x_in = (const float*)d_in[0];   // [B, D, T] fp32
    const float* emb  = (const float*)d_in[1];   // [K, D] fp32

    float* out      = (float*)d_out;
    float* out_q    = out;                                   // 16,777,216 floats
    float* out_loss = out + (size_t)BATCH * EMB_DIM * T_SZ;  // 1 float
    float* out_idx  = out_loss + 1;                          // 65,536 floats

    // Workspace: e_hi (512KB) | e_lo (512KB, MUST follow e_hi: TDM plane 2) |
    //            half_norm (4KB)
    _Float16* e_hi = (_Float16*)d_ws;
    _Float16* e_lo = e_hi + (size_t)NUM_EMB * EMB_DIM;
    float* half_norm = (float*)(e_lo + (size_t)NUM_EMB * EMB_DIM);

    hipMemsetAsync(out_loss, 0, sizeof(float), stream);   // graph-capturable

    vq_prep_kernel<<<NUM_EMB, EMB_DIM, 0, stream>>>(emb, e_hi, e_lo, half_norm);
    vq_main_kernel<<<N_ROWS / 128, 256, 0, stream>>>(x_in, emb, e_hi, half_norm,
                                                     out_q, out_loss, out_idx);
}
